// Attn_79671643341302
// MI455X (gfx1250) — matrix operand order repaired and checked by reading, not compiled, not run on hardware
//
#include <hip/hip_runtime.h>
#include <hip/hip_bf16.h>
#include <math.h>

// ---------------------------------------------------------------------------
// Types for CDNA5 WMMA (wave32): 16x16x32 bf16 -> f32
// ---------------------------------------------------------------------------
typedef __attribute__((ext_vector_type(16))) __bf16 v16bf;
typedef __attribute__((ext_vector_type(8)))  __bf16 v8bf;
typedef __attribute__((ext_vector_type(8)))  float  v8f;

#define D_MODEL 2048
#define KV_DIM  512
#define N_HEADS 16
#define N_KV    4
#define HEAD_DIM 128

__device__ __forceinline__ v8f wmma_bf16(v16bf a, v16bf b, v8f c) {
    return __builtin_amdgcn_wmma_f32_16x16x32_bf16(
        /*neg_a=*/false, a, /*neg_b=*/false, b,
        /*c_mod=*/(short)0, c, /*reuse_a=*/false, /*reuse_b=*/false);
}

__device__ __forceinline__ v16bf cat16(v8bf lo, v8bf hi) {
    return __builtin_shufflevector(lo, hi,
        0,1,2,3,4,5,6,7,8,9,10,11,12,13,14,15);
}

// ---------------------------------------------------------------------------
// f32 -> bf16 conversion
// ---------------------------------------------------------------------------
__global__ void to_bf16_kernel(const float* __restrict__ in,
                               __bf16* __restrict__ out, int n) {
    int i = blockIdx.x * 256 + threadIdx.x;
    if (i < n) out[i] = (__bf16)in[i];
}

// ---------------------------------------------------------------------------
// Block-tiled GEMM: C(MxN, f32) = A(MxK, bf16 row-major) * W(NxK, bf16)^T
//
// Block = 256 threads (8 waves) owns a 128x64 C tile; each wave a 32x32
// sub-tile (4 wmma accumulators). A/B tiles (BK=64) are staged in LDS via
// CDNA5 async global->LDS copies (ASYNCcnt), double-buffered so the next
// tile's fill overlaps the current tile's 8 WMMAs per wave.
// LDS rows padded to 72 bf16 (36 dwords) to avoid bank conflicts.
// ---------------------------------------------------------------------------
#define BM 128
#define BN 64
#define BK 64
#define LPAD 72

__global__ __launch_bounds__(256)
void gemm_bt_kernel(const __bf16* __restrict__ A, const __bf16* __restrict__ W,
                    float* __restrict__ C, int M, int N, int K) {
    __shared__ __align__(16) __bf16 As[2][BM][LPAD];
    __shared__ __align__(16) __bf16 Bs[2][BN][LPAD];

    const int m0   = blockIdx.y * BM;
    const int n0   = blockIdx.x * BN;
    const int tid  = threadIdx.x;
    const int wid  = tid >> 5;
    const int lane = tid & 31;
    const int mw   = wid >> 1;          // 0..3 : 32-row band
    const int nw   = wid & 1;           // 0..1 : 32-col band
    const int nn   = lane & 15;
    const int lh   = lane >> 4;

    // Issue one BMxBK + BNxBK tile fill as async 16B/lane copies.
    auto issue_tile = [&](int buf, int k0) {
        #pragma unroll
        for (int rep = 0; rep < 4; ++rep) {           // A: 1024 chunks / 256 thr
            int ci  = tid + rep * 256;
            int row = ci >> 3;
            int col = (ci & 7) * 8;
            const __bf16* g = A + (size_t)(m0 + row) * K + k0 + col;
            unsigned lo = (unsigned)(unsigned long long)&As[buf][row][col];
            asm volatile("global_load_async_to_lds_b128 %0, %1, off"
                         :: "v"(lo), "v"((unsigned long long)g) : "memory");
        }
        #pragma unroll
        for (int rep = 0; rep < 2; ++rep) {           // B: 512 chunks / 256 thr
            int ci  = tid + rep * 256;
            int row = ci >> 3;
            int col = (ci & 7) * 8;
            const __bf16* g = W + (size_t)(n0 + row) * K + k0 + col;
            unsigned lo = (unsigned)(unsigned long long)&Bs[buf][row][col];
            asm volatile("global_load_async_to_lds_b128 %0, %1, off"
                         :: "v"(lo), "v"((unsigned long long)g) : "memory");
        }
    };

    v8f acc00 = {}, acc01 = {}, acc10 = {}, acc11 = {};

    issue_tile(0, 0);
    asm volatile("s_wait_asynccnt 0" ::: "memory");
    __syncthreads();

    const int ntile = K / BK;
    for (int kt = 0; kt < ntile; ++kt) {
        const int cur = kt & 1;
        if (kt + 1 < ntile) issue_tile(cur ^ 1, (kt + 1) * BK);

        const int am = mw * 32;
        const int bn = nw * 32;
        #pragma unroll
        for (int ks = 0; ks < BK; ks += 32) {
            // A fragments (16-bit A layout: two contiguous 16B runs per lane)
            v16bf a0 = cat16(*(const v8bf*)&As[cur][am + nn][ks + lh * 8],
                             *(const v8bf*)&As[cur][am + nn][ks + 16 + lh * 8]);
            v16bf a1 = cat16(*(const v8bf*)&As[cur][am + 16 + nn][ks + lh * 8],
                             *(const v8bf*)&As[cur][am + 16 + nn][ks + 16 + lh * 8]);
            // B fragments (same k order as A: two 16B runs per lane)
            v16bf b0 = cat16(*(const v8bf*)&Bs[cur][bn + nn][ks + lh * 8],
                             *(const v8bf*)&Bs[cur][bn + nn][ks + 16 + lh * 8]);
            v16bf b1 = cat16(*(const v8bf*)&Bs[cur][bn + 16 + nn][ks + lh * 8],
                             *(const v8bf*)&Bs[cur][bn + 16 + nn][ks + 16 + lh * 8]);
            acc00 = wmma_bf16(a0, b0, acc00);
            acc01 = wmma_bf16(a0, b1, acc01);
            acc10 = wmma_bf16(a1, b0, acc10);
            acc11 = wmma_bf16(a1, b1, acc11);
        }

        asm volatile("s_wait_asynccnt 0" ::: "memory");
        __syncthreads();
    }

    const int h8 = lh * 8;
    #pragma unroll
    for (int r = 0; r < 8; ++r) {
        size_t row0 = (size_t)(m0 + mw * 32 + h8 + r);
        size_t row1 = row0 + 16;
        C[row0 * N + n0 + nw * 32 + nn]      = acc00[r];
        C[row0 * N + n0 + nw * 32 + 16 + nn] = acc01[r];
        C[row1 * N + n0 + nw * 32 + nn]      = acc10[r];
        C[row1 * N + n0 + nw * 32 + 16 + nn] = acc11[r];
    }
}

// ---------------------------------------------------------------------------
// RMSNorm + RoPE (+ q-gain) over one head of one token. f32 in -> bf16 out.
// grid: (b*s, NH + NKV), block: 128 threads (= head_dim).
// ---------------------------------------------------------------------------
__global__ __launch_bounds__(128)
void norm_rope_kernel(const float* __restrict__ qf, const float* __restrict__ kf,
                      const float* __restrict__ qg,
                      __bf16* __restrict__ qb, __bf16* __restrict__ kb, int S) {
    const float eps = 1.1920929e-7f;
    int row  = blockIdx.x;          // b*s token index
    int head = blockIdx.y;          // 0..NH-1 => Q, NH..NH+NKV-1 => K
    int t    = threadIdx.x;         // 0..127

    const float* src;
    __bf16* dst;
    float gain = 1.0f;
    if (head < N_HEADS) {
        src  = qf + (size_t)row * D_MODEL + head * HEAD_DIM;
        dst  = qb + (size_t)row * D_MODEL + head * HEAD_DIM;
        gain = qg[head];
    } else {
        int hk = head - N_HEADS;
        src = kf + (size_t)row * KV_DIM + hk * HEAD_DIM;
        dst = kb + (size_t)row * KV_DIM + hk * HEAD_DIM;
    }

    float v  = src[t];
    float ss = v * v;
    #pragma unroll
    for (int off = 1; off < 32; off <<= 1) ss += __shfl_xor(ss, off, 32);

    __shared__ float red[4];
    int wv = t >> 5;
    if ((t & 31) == 0) red[wv] = ss;
    __syncthreads();
    float total = red[0] + red[1] + red[2] + red[3];
    float rms   = rsqrtf(total * (1.0f / HEAD_DIM) + eps);

    float xn = v * rms;
    float pn = src[t ^ 64] * rms;   // rotation partner

    int   pos   = row % S;
    int   i     = t & 63;
    float freq  = __expf(-(float)(2 * i) * (1.0f / HEAD_DIM) * logf(10000.0f));
    float ang   = (float)pos * freq;
    float c     = cosf(ang);
    float snv   = sinf(ang);

    float o = (t < 64) ? (xn * c + pn * snv) : (-pn * snv + xn * c);
    dst[t] = (__bf16)(o * gain);
}

// ---------------------------------------------------------------------------
// V transpose: vf (b*s, kv) f32 -> vt[b][hkv][hd][s] bf16 (so P@V B-frags
// are two contiguous 16B loads along the key axis).
// ---------------------------------------------------------------------------
__global__ void vtrans_kernel(const float* __restrict__ vf,
                              __bf16* __restrict__ vt, int S, int n) {
    int i = blockIdx.x * 256 + threadIdx.x;
    if (i >= n) return;
    int row = i / KV_DIM;           // b*s
    int c0  = i % KV_DIM;
    int bi  = row / S;
    int pos = row % S;
    int hkv = c0 / HEAD_DIM;
    int c   = c0 % HEAD_DIM;
    vt[((size_t)(bi * N_KV + hkv) * HEAD_DIM + c) * S + pos] = (__bf16)vf[i];
}

// ---------------------------------------------------------------------------
// Causal flash attention. One wave per (b, h, 16-query tile).
// S-tiles via 2x(4 wmma), online softmax with 16-lane shfl reductions,
// P transposed through per-wave LDS staging, then 8 wmma for P@V.
// ---------------------------------------------------------------------------
__global__ __launch_bounds__(256)
void flash_kernel(const __bf16* __restrict__ qb, const __bf16* __restrict__ kb,
                  const __bf16* __restrict__ vt, __bf16* __restrict__ yb,
                  int B, int S) {
    __shared__ __align__(32) __bf16 Pbuf[8][16][32];

    int wv   = threadIdx.x >> 5;
    int lane = threadIdx.x & 31;
    int wid  = blockIdx.x * 8 + wv;

    int qtiles = S >> 4;
    int qt = wid % qtiles;
    int h  = (wid / qtiles) % N_HEADS;
    int bi = wid / (qtiles * N_HEADS);
    if (bi >= B) return;

    int q0  = qt * 16;
    int hkv = h >> 2;                 // NH/NKV = 4
    int nn  = lane & 15;
    int lh  = lane >> 4;

    // Q fragments: 4 chunks of K-dim 32 each (hd = 128)
    v16bf qfr[4];
    #pragma unroll
    for (int c = 0; c < 4; ++c) {
        const __bf16* p = qb + (size_t)(bi * S + q0 + nn) * D_MODEL
                             + h * HEAD_DIM + c * 32 + lh * 8;
        qfr[c] = cat16(*(const v8bf*)p, *(const v8bf*)(p + 16));
    }

    float mi[8], li[8];
    v8f acc[8];
    #pragma unroll
    for (int r = 0; r < 8; ++r) { mi[r] = -3.0e38f; li[r] = 0.0f; }
    #pragma unroll
    for (int cc = 0; cc < 8; ++cc) acc[cc] = (v8f){};

    const float scale = 0.08838834764831845f;  // 1/sqrt(128)
    const __bf16* Kb  = kb + (size_t)(bi * S) * KV_DIM;
    const __bf16* Vtb = vt + (size_t)(bi * N_KV + hkv) * HEAD_DIM * S;

    for (int k0 = 0; k0 < q0 + 16; k0 += 32) {
        // ---- S = Q K^T for 32 keys (two 16-col tiles)
        v8f s0 = (v8f){}, s1 = (v8f){};
        #pragma unroll
        for (int c = 0; c < 4; ++c) {
            const __bf16* kp0 = Kb + (size_t)(k0 + nn) * KV_DIM
                                   + hkv * HEAD_DIM + c * 32 + lh * 8;
            const __bf16* kp1 = Kb + (size_t)(k0 + 16 + nn) * KV_DIM
                                   + hkv * HEAD_DIM + c * 32 + lh * 8;
            v16bf b0 = cat16(*(const v8bf*)kp0, *(const v8bf*)(kp0 + 16));
            v16bf b1 = cat16(*(const v8bf*)kp1, *(const v8bf*)(kp1 + 16));
            s0 = wmma_bf16(qfr[c], b0, s0);
            s1 = wmma_bf16(qfr[c], b1, s1);
        }

        // ---- online softmax (row stats reduced across the 16-lane half)
        float alpha[8], p0[8], p1[8];
        #pragma unroll
        for (int r = 0; r < 8; ++r) {
            int qrow = q0 + lh * 8 + r;
            float t0 = (k0 + nn      <= qrow) ? s0[r] * scale : -3.0e38f;
            float t1 = (k0 + 16 + nn <= qrow) ? s1[r] * scale : -3.0e38f;
            float mx = fmaxf(t0, t1);
            #pragma unroll
            for (int off = 1; off < 16; off <<= 1)
                mx = fmaxf(mx, __shfl_xor(mx, off, 32));
            float mnew = fmaxf(mi[r], mx);
            alpha[r] = __expf(mi[r] - mnew);
            p0[r] = __expf(t0 - mnew);
            p1[r] = __expf(t1 - mnew);
            float ps = p0[r] + p1[r];
            #pragma unroll
            for (int off = 1; off < 16; off <<= 1)
                ps += __shfl_xor(ps, off, 32);
            li[r] = li[r] * alpha[r] + ps;
            mi[r] = mnew;
        }

        // ---- transpose P (D-layout -> A-layout) via private LDS tile
        #pragma unroll
        for (int r = 0; r < 8; ++r) {
            Pbuf[wv][lh * 8 + r][nn]      = (__bf16)p0[r];
            Pbuf[wv][lh * 8 + r][16 + nn] = (__bf16)p1[r];
        }
        asm volatile("s_wait_dscnt 0" ::: "memory");

        v16bf pf = cat16(*(const v8bf*)&Pbuf[wv][nn][lh * 8],
                         *(const v8bf*)&Pbuf[wv][nn][16 + lh * 8]);

        // ---- O = alpha*O + P @ V  (8 hd-chunks of 16 cols)
        #pragma unroll
        for (int cc = 0; cc < 8; ++cc) {
            const __bf16* vp = Vtb + (size_t)(cc * 16 + nn) * S + k0 + lh * 8;
            v16bf vfrag = cat16(*(const v8bf*)vp, *(const v8bf*)(vp + 16));
            v8f c = acc[cc];
            #pragma unroll
            for (int r = 0; r < 8; ++r) c[r] *= alpha[r];
            acc[cc] = wmma_bf16(pf, vfrag, c);
        }
    }

    // ---- normalize and store (bf16 input for the out-projection GEMM)
    #pragma unroll
    for (int cc = 0; cc < 8; ++cc) {
        #pragma unroll
        for (int r = 0; r < 8; ++r) {
            int qrow = q0 + lh * 8 + r;
            float o  = acc[cc][r] / li[r];
            yb[(size_t)(bi * S + qrow) * D_MODEL + h * HEAD_DIM + cc * 16 + nn]
                = (__bf16)o;
        }
    }
}

// ---------------------------------------------------------------------------
// Host-side orchestration
// ---------------------------------------------------------------------------
extern "C" void kernel_launch(void* const* d_in, const int* in_sizes, int n_in,
                              void* d_out, int out_size, void* d_ws, size_t ws_size,
                              hipStream_t stream) {
    const float* x  = (const float*)d_in[0];
    const float* Wq = (const float*)d_in[1];
    const float* Wk = (const float*)d_in[2];
    const float* Wv = (const float*)d_in[3];
    const float* Wo = (const float*)d_in[4];
    const float* qg = (const float*)d_in[5];

    const int S = 2048;
    const int B = in_sizes[0] / (S * D_MODEL);
    const int M = B * S;

    // Workspace carve-up (256B aligned)
    char* wp = (char*)d_ws;
    auto alloc = [&](size_t bytes) -> void* {
        void* p = (void*)wp;
        wp += (bytes + 255) & ~(size_t)255;
        return p;
    };
    __bf16* xb  = (__bf16*)alloc((size_t)M * D_MODEL * 2);
    __bf16* wqb = (__bf16*)alloc((size_t)D_MODEL * D_MODEL * 2);
    __bf16* wkb = (__bf16*)alloc((size_t)KV_DIM * D_MODEL * 2);
    __bf16* wvb = (__bf16*)alloc((size_t)KV_DIM * D_MODEL * 2);
    __bf16* wob = (__bf16*)alloc((size_t)D_MODEL * D_MODEL * 2);
    float*  qf  = (float*) alloc((size_t)M * D_MODEL * 4);
    float*  kf  = (float*) alloc((size_t)M * KV_DIM * 4);
    float*  vf  = (float*) alloc((size_t)M * KV_DIM * 4);
    __bf16* qbn = (__bf16*)alloc((size_t)M * D_MODEL * 2);
    __bf16* kbn = (__bf16*)alloc((size_t)M * KV_DIM * 2);
    __bf16* vtb = (__bf16*)alloc((size_t)M * KV_DIM * 2);
    __bf16* yb  = (__bf16*)alloc((size_t)M * D_MODEL * 2);

    // 1) f32 -> bf16
    {
        int n;
        n = M * D_MODEL;
        to_bf16_kernel<<<(n + 255) / 256, 256, 0, stream>>>(x, xb, n);
        n = D_MODEL * D_MODEL;
        to_bf16_kernel<<<(n + 255) / 256, 256, 0, stream>>>(Wq, wqb, n);
        n = KV_DIM * D_MODEL;
        to_bf16_kernel<<<(n + 255) / 256, 256, 0, stream>>>(Wk, wkb, n);
        to_bf16_kernel<<<(n + 255) / 256, 256, 0, stream>>>(Wv, wvb, n);
        n = D_MODEL * D_MODEL;
        to_bf16_kernel<<<(n + 255) / 256, 256, 0, stream>>>(Wo, wob, n);
    }

    // 2) QKV projections (LDS-tiled async WMMA GEMMs)
    {
        dim3 gq(D_MODEL / BN, M / BM);
        gemm_bt_kernel<<<gq, 256, 0, stream>>>(xb, wqb, qf, M, D_MODEL, D_MODEL);
        dim3 gkv(KV_DIM / BN, M / BM);
        gemm_bt_kernel<<<gkv, 256, 0, stream>>>(xb, wkb, kf, M, KV_DIM, D_MODEL);
        gemm_bt_kernel<<<gkv, 256, 0, stream>>>(xb, wvb, vf, M, KV_DIM, D_MODEL);
    }

    // 3) RMSNorm + RoPE (+ gain) and V transpose
    {
        dim3 grid(M, N_HEADS + N_KV);
        norm_rope_kernel<<<grid, 128, 0, stream>>>(qf, kf, qg, qbn, kbn, S);
        int n = M * KV_DIM;
        vtrans_kernel<<<(n + 255) / 256, 256, 0, stream>>>(vf, vtb, S, n);
    }

    // 4) Causal flash attention
    {
        int waves = B * N_HEADS * (S / 16);
        flash_kernel<<<waves / 8, 256, 0, stream>>>(qbn, kbn, vtb, yb, B, S);
    }

    // 5) Output projection -> f32 d_out
    {
        dim3 go(D_MODEL / BN, M / BM);
        gemm_bt_kernel<<<go, 256, 0, stream>>>(yb, wob, (float*)d_out,
                                               M, D_MODEL, D_MODEL);
    }
}
